// CausalSelfAttention_75780402971163
// MI455X (gfx1250) — matrix operand order repaired and checked by reading, not compiled, not run on hardware
//
#include <hip/hip_runtime.h>

// ---------------------------------------------------------------------------
// Causal self-attention forward for MI455X (gfx1250, wave32, WMMA + TDM).
// B=2, T=2048, C=1024, H=16, D=64. All matrix math on v_wmma_f32_16x16x32_bf16.
// GEMM tiles staged into LDS with the Tensor Data Mover (tensor_load_to_lds),
// double-buffered, synchronized with s_wait_tensorcnt + workgroup barriers.
// ---------------------------------------------------------------------------

typedef __attribute__((ext_vector_type(16))) __bf16        v16bf;
typedef __attribute__((ext_vector_type(8)))  float         v8f;
typedef __attribute__((ext_vector_type(4)))  unsigned int  v4u;
typedef __attribute__((ext_vector_type(8)))  int           v8i;
typedef __attribute__((ext_vector_type(4)))  int           v4i;

#define B_ 2
#define T_ 2048
#define C_ 1024
#define H_ 16
#define D_ 64
#define EPS_ 1.1920928955078125e-07f

#if __has_builtin(__builtin_amdgcn_tensor_load_to_lds) && \
    __has_builtin(__builtin_amdgcn_s_wait_tensorcnt)
#define HAVE_TDM 1
#else
#define HAVE_TDM 0
#endif

// ------------------------------- helpers -----------------------------------

__device__ __forceinline__ float halfwave_max(float v) {
  #pragma unroll
  for (int m = 1; m < 16; m <<= 1) v = fmaxf(v, __shfl_xor(v, m, 32));
  return v;
}
__device__ __forceinline__ float halfwave_sum(float v) {
  #pragma unroll
  for (int m = 1; m < 16; m <<= 1) v += __shfl_xor(v, m, 32);
  return v;
}

#if HAVE_TDM
// Issue one 2D TDM tile load: tile_rows x tile_cols bf16 elements starting at
// gptr (row length = row_stride elements), into LDS at byte offset lds_addr.
// D# packing per CDNA5 ISA sec. 8 (groups 0/1; groups 2/3 zero for 2D).
// This toolchain exposes the 6-arg builtin:
//   (uint32x4 g0, int32x8 g1, int32x4 g2, int32x4 g3, int32x8 g4, i32 cpol)
__device__ __forceinline__ void tdm_load_2d(unsigned lds_addr,
                                            const __bf16* gptr,
                                            unsigned tensor_rows,
                                            unsigned row_stride,
                                            unsigned tile_cols,
                                            unsigned tile_rows) {
  const unsigned long long ga = (unsigned long long)(uintptr_t)gptr;
  v4u g0;
  g0[0] = 1u;                                   // count=1 (valid), user mode
  g0[1] = lds_addr;                             // lds_addr[31:0]
  g0[2] = (unsigned)ga;                         // global_addr[31:0]
  g0[3] = (unsigned)((ga >> 32) & 0x1FFFFFFu)   // global_addr[56:32]
        | (2u << 30);                           // type = 2 ("image")
  v8i g1;
  g1[0] = (int)(1u << 16);                      // data_size=1 -> 2 bytes
  g1[1] = (int)((row_stride & 0xFFFFu) << 16);  // tensor_dim0[15:0]
  g1[2] = (int)((row_stride >> 16) | ((tensor_rows & 0xFFFFu) << 16));
  g1[3] = (int)((tensor_rows >> 16) | (tile_cols << 16));   // tile_dim0
  g1[4] = (int)tile_rows;                       // tile_dim1 (tile_dim2=0)
  g1[5] = (int)row_stride;                      // tensor_dim0_stride[31:0]
  g1[6] = 0;
  g1[7] = 0;
  v4i z4 = {0, 0, 0, 0};
  v8i z8 = {0, 0, 0, 0, 0, 0, 0, 0};
  __builtin_amdgcn_tensor_load_to_lds(g0, g1, z4, z4, z8, 0);
}
#endif

// --------------------------- 1) f32 -> bf16 --------------------------------

__global__ void f32_to_bf16(const float* __restrict__ s, __bf16* __restrict__ d, int n) {
  int i = blockIdx.x * blockDim.x + threadIdx.x;
  int stride = gridDim.x * blockDim.x;
  for (; i < n; i += stride) d[i] = (__bf16)s[i];
}

// --------------------- 2) GEMM  Y = A(MxK) @ W(NxK)^T ----------------------
// Workgroup = 8 waves as 4(M) x 2(N); wave tile 32x64 = 2x4 accumulators;
// workgroup tile 128x128. K-step 64, double-buffered LDS staged by the TDM.
// Fragment layouts used here: A lane -> row M=lane%16, 16 contiguous K at
// (lane>=16 ? 16 : 0). B lane -> col n=lane%16, 16 contiguous K at
// (lane>=16 ? 16 : 0). Both contiguous in (LDS) memory.
__global__ __launch_bounds__(256) void gemm_bf16_wmma(
    const __bf16* __restrict__ A,  // M x K row-major
    const __bf16* __restrict__ W,  // N x K row-major (acts as B = W^T)
    float* __restrict__ Y,         // M x N row-major
    int M, int N, int K) {
  const int lane = threadIdx.x & 31;
  const int wave = threadIdx.x >> 5;
  const int hi = lane >> 4;
  const int l15 = lane & 15;
  const int wm = wave >> 1;       // 0..3
  const int wn = wave & 1;        // 0..1
  const int m0 = blockIdx.x * 128;
  const int n0 = blockIdx.y * 128;

  v8f acc[2][4] = {};

#if HAVE_TDM
  __shared__ __bf16 smA[2][128 * 64];
  __shared__ __bf16 smB[2][128 * 64];
  const unsigned ldsA0 = (unsigned)(uintptr_t)&smA[0][0];
  const unsigned ldsA1 = (unsigned)(uintptr_t)&smA[1][0];
  const unsigned ldsB0 = (unsigned)(uintptr_t)&smB[0][0];
  const unsigned ldsB1 = (unsigned)(uintptr_t)&smB[1][0];

  if (wave == 0) {
    tdm_load_2d(ldsA0, A + (size_t)m0 * K, (unsigned)M, (unsigned)K, 64u, 128u);
    tdm_load_2d(ldsB0, W + (size_t)n0 * K, (unsigned)N, (unsigned)K, 64u, 128u);
  }

  for (int k = 0, buf = 0; k < K; k += 64, buf ^= 1) {
    if (wave == 0) __builtin_amdgcn_s_wait_tensorcnt(0);
    __syncthreads();
    if (wave == 0 && (k + 64) < K) {
      tdm_load_2d(buf ? ldsA0 : ldsA1, A + (size_t)m0 * K + k + 64,
                  (unsigned)M, (unsigned)K, 64u, 128u);
      tdm_load_2d(buf ? ldsB0 : ldsB1, W + (size_t)n0 * K + k + 64,
                  (unsigned)N, (unsigned)K, 64u, 128u);
    }
    const __bf16* pa = &smA[buf][0];
    const __bf16* pb = &smB[buf][0];
    #pragma unroll
    for (int kc = 0; kc < 64; kc += 32) {
      v16bf a[2], b[4];
      #pragma unroll
      for (int i = 0; i < 2; ++i) {
        const __bf16* p = pa + (size_t)(wm * 32 + i * 16 + l15) * 64 + kc + hi * 16;
        #pragma unroll
        for (int t = 0; t < 8; ++t) { a[i][t] = p[t]; a[i][8 + t] = p[8 + t]; }
      }
      #pragma unroll
      for (int j = 0; j < 4; ++j) {
        const __bf16* p = pb + (size_t)(wn * 64 + j * 16 + l15) * 64 + kc + hi * 16;
        #pragma unroll
        for (int t = 0; t < 16; ++t) b[j][t] = p[t];
      }
      #pragma unroll
      for (int i = 0; i < 2; ++i)
        #pragma unroll
        for (int j = 0; j < 4; ++j)
          acc[i][j] = __builtin_amdgcn_wmma_f32_16x16x32_bf16(
              false, a[i], false, b[j], (short)0, acc[i][j], false, false);
    }
  }
#else
  // Fallback: direct global fragment loads.
  const __bf16* pa0 = A + (size_t)(m0 + wm * 32 + l15) * K;
  const __bf16* pa1 = pa0 + (size_t)16 * K;
  for (int k = 0; k < K; k += 32) {
    v16bf a[2], b[4];
    __builtin_prefetch(pa0 + k + 128, 0, 1);
    #pragma unroll
    for (int i = 0; i < 2; ++i) {
      const __bf16* p = (i ? pa1 : pa0) + k + hi * 16;
      #pragma unroll
      for (int t = 0; t < 8; ++t) { a[i][t] = p[t]; a[i][8 + t] = p[8 + t]; }
    }
    #pragma unroll
    for (int j = 0; j < 4; ++j) {
      const __bf16* p = W + (size_t)(n0 + wn * 64 + j * 16 + l15) * K + k + hi * 16;
      #pragma unroll
      for (int t = 0; t < 16; ++t) b[j][t] = p[t];
    }
    #pragma unroll
    for (int i = 0; i < 2; ++i)
      #pragma unroll
      for (int j = 0; j < 4; ++j)
        acc[i][j] = __builtin_amdgcn_wmma_f32_16x16x32_bf16(
            false, a[i], false, b[j], (short)0, acc[i][j], false, false);
  }
#endif

  // D fragment: element r of lane -> (M = tile + r + 8*hi, N = tile + lane%16)
  float* py = Y + (size_t)(m0 + wm * 32 + hi * 8) * N + n0 + wn * 64 + l15;
  #pragma unroll
  for (int i = 0; i < 2; ++i)
    #pragma unroll
    for (int r = 0; r < 8; ++r)
      #pragma unroll
      for (int j = 0; j < 4; ++j)
        py[(size_t)(i * 16 + r) * N + j * 16] = acc[i][j][r];
}

// ----------------- 3a) RMSNorm + RoPE for q/k, relayout --------------------
// In:  X  (B,T,H,D) f32   Out: O (B,H,T,D) bf16. One wave per head-row.
__global__ __launch_bounds__(256) void norm_rope_kernel(
    const float* __restrict__ X, const float* __restrict__ w,
    __bf16* __restrict__ O) {
  const int wave = threadIdx.x >> 5;
  const int lane = threadIdx.x & 31;
  const int row = blockIdx.x * 8 + wave;      // (b*T + t)*H + h
  const int h = row & (H_ - 1);
  const int bt = row >> 4;
  const int t = bt & (T_ - 1);
  const int b = bt >> 11;

  const float* px = X + (size_t)row * D_;
  float x0 = px[lane], x1 = px[lane + 32];
  float ss = x0 * x0 + x1 * x1;
  #pragma unroll
  for (int m = 1; m < 32; m <<= 1) ss += __shfl_xor(ss, m, 32);
  const float inv = rsqrtf(ss * (1.0f / D_) + EPS_);
  x0 = x0 * inv * w[lane];
  x1 = x1 * inv * w[lane + 32];

  // RoPE: lane i holds the pair (x[i], x[i+32]); inv_freq = 10000^{-i/32}
  const float fr = __powf(10000.0f, -(float)lane * (1.0f / 32.0f));
  const float ang = (float)t * fr;
  float sn, cs;
  __sincosf(ang, &sn, &cs);
  const float o0 = x0 * cs - x1 * sn;
  const float o1 = x0 * sn + x1 * cs;

  __bf16* po = O + ((size_t)(b * H_ + h) * T_ + t) * D_;
  po[lane] = (__bf16)o0;
  po[lane + 32] = (__bf16)o1;
}

// -------------- 3b) V: (B,T,H,D) f32 -> Vt (B,H,D,T) bf16 ------------------
__global__ void v_transpose_kernel(const float* __restrict__ V,
                                   __bf16* __restrict__ Vt) {
  const int i = blockIdx.x * 256 + threadIdx.x;  // one per element
  const int d = i & (D_ - 1);
  const int h = (i >> 6) & (H_ - 1);
  const int bt = i >> 10;                        // b*T + t
  const int t = bt & (T_ - 1);
  const int b = bt >> 11;
  Vt[((size_t)(b * H_ + h) * D_ + d) * T_ + t] = (__bf16)V[i];
}

// ----------------------- 4) flash attention per head -----------------------
// Grid: (T/128, B*H). 8 waves/block; wave w owns Q rows q0 = bx*128 + w*16.
// Loop KV tiles of 32 (causal). S via 2 WMMAs (D=64), online softmax with
// half-wave shuffles, P -> LDS -> A-fragment, O += P@V via 4 WMMAs.
__global__ __launch_bounds__(256) void flash_attn_kernel(
    const __bf16* __restrict__ Q,   // (B,H,T,D)
    const __bf16* __restrict__ Kk,  // (B,H,T,D)
    const __bf16* __restrict__ Vt,  // (B,H,D,T)
    __bf16* __restrict__ O) {       // (B,T,C)
  __shared__ __bf16 plds[8][16 * 32];

  const int wave = threadIdx.x >> 5;
  const int lane = threadIdx.x & 31;
  const int hi = lane >> 4;
  const int l15 = lane & 15;
  const int bh = blockIdx.y;  // b*H + h
  const int b = bh >> 4;
  const int h = bh & (H_ - 1);
  const int q0 = blockIdx.x * 128 + wave * 16;

  const __bf16* qbase = Q + ((size_t)bh * T_ + q0) * D_;
  const __bf16* kbase = Kk + (size_t)bh * T_ * D_;
  const __bf16* vbase = Vt + (size_t)bh * D_ * T_;

  // Q A-fragments for the two K=32 chunks of D=64 (persist across loop)
  v16bf aq0, aq1;
  {
    const __bf16* p0 = qbase + (size_t)l15 * D_ + 0 + hi * 16;
    const __bf16* p1 = qbase + (size_t)l15 * D_ + 32 + hi * 16;
    #pragma unroll
    for (int i = 0; i < 8; ++i) {
      aq0[i] = p0[i]; aq0[8 + i] = p0[8 + i];
      aq1[i] = p1[i]; aq1[8 + i] = p1[8 + i];
    }
  }

  float mrow[8], lrow[8];
  v8f o0 = {}, o1 = {}, o2 = {}, o3 = {};
  #pragma unroll
  for (int r = 0; r < 8; ++r) { mrow[r] = -INFINITY; lrow[r] = 0.0f; }

  const int qmax = q0 + 15;
  const float scale = 0.125f;  // 1/sqrt(64)

  for (int j0 = 0; j0 <= qmax; j0 += 32) {
    // ---- S = Q @ K^T for 32 keys (two 16-key subtiles) ----
    v8f s0 = {}, s1 = {};
    {
      v16bf bk;
      const __bf16* p;
      p = kbase + (size_t)(j0 + l15) * D_ + 0 + hi * 16;
      #pragma unroll
      for (int i = 0; i < 16; ++i) bk[i] = p[i];
      s0 = __builtin_amdgcn_wmma_f32_16x16x32_bf16(false, aq0, false, bk, (short)0, s0, false, false);
      p = kbase + (size_t)(j0 + l15) * D_ + 32 + hi * 16;
      #pragma unroll
      for (int i = 0; i < 16; ++i) bk[i] = p[i];
      s0 = __builtin_amdgcn_wmma_f32_16x16x32_bf16(false, aq1, false, bk, (short)0, s0, false, false);
      p = kbase + (size_t)(j0 + 16 + l15) * D_ + 0 + hi * 16;
      #pragma unroll
      for (int i = 0; i < 16; ++i) bk[i] = p[i];
      s1 = __builtin_amdgcn_wmma_f32_16x16x32_bf16(false, aq0, false, bk, (short)0, s1, false, false);
      p = kbase + (size_t)(j0 + 16 + l15) * D_ + 32 + hi * 16;
      #pragma unroll
      for (int i = 0; i < 16; ++i) bk[i] = p[i];
      s1 = __builtin_amdgcn_wmma_f32_16x16x32_bf16(false, aq1, false, bk, (short)0, s1, false, false);
    }

    // ---- scale, causal mask, online softmax ----
    #pragma unroll
    for (int r = 0; r < 8; ++r) {
      const int qrow = q0 + r + hi * 8;
      float e0 = s0[r] * scale;
      float e1 = s1[r] * scale;
      if (j0 + l15 > qrow) e0 = -INFINITY;
      if (j0 + 16 + l15 > qrow) e1 = -INFINITY;
      float mx = halfwave_max(fmaxf(e0, e1));
      const float mnew = fmaxf(mrow[r], mx);
      const float alpha = __expf(mrow[r] - mnew);
      const float p0 = __expf(e0 - mnew);
      const float p1 = __expf(e1 - mnew);
      const float rs = halfwave_sum(p0 + p1);
      lrow[r] = lrow[r] * alpha + rs;
      mrow[r] = mnew;
      o0[r] *= alpha; o1[r] *= alpha; o2[r] *= alpha; o3[r] *= alpha;
      const int rowb = r + hi * 8;
      plds[wave][rowb * 32 + l15] = (__bf16)p0;
      plds[wave][rowb * 32 + 16 + l15] = (__bf16)p1;
    }

#if __has_builtin(__builtin_amdgcn_s_wait_dscnt)
    __builtin_amdgcn_s_wait_dscnt(0);
#else
    asm volatile("s_wait_dscnt 0" ::: "memory");
#endif
    __builtin_amdgcn_wave_barrier();

    // ---- read P back as an A-fragment (16x32) ----
    v16bf ap;
    {
      const __bf16* pp = &plds[wave][l15 * 32 + hi * 16];
      #pragma unroll
      for (int i = 0; i < 8; ++i) { ap[i] = pp[i]; ap[8 + i] = pp[8 + i]; }
    }

    // ---- O += P @ V : 4 column tiles of D ----
    {
      v16bf bv;
      const __bf16* p;
      p = vbase + (size_t)(0 + l15) * T_ + j0 + hi * 16;
      #pragma unroll
      for (int i = 0; i < 16; ++i) bv[i] = p[i];
      o0 = __builtin_amdgcn_wmma_f32_16x16x32_bf16(false, ap, false, bv, (short)0, o0, false, false);
      p = vbase + (size_t)(16 + l15) * T_ + j0 + hi * 16;
      #pragma unroll
      for (int i = 0; i < 16; ++i) bv[i] = p[i];
      o1 = __builtin_amdgcn_wmma_f32_16x16x32_bf16(false, ap, false, bv, (short)0, o1, false, false);
      p = vbase + (size_t)(32 + l15) * T_ + j0 + hi * 16;
      #pragma unroll
      for (int i = 0; i < 16; ++i) bv[i] = p[i];
      o2 = __builtin_amdgcn_wmma_f32_16x16x32_bf16(false, ap, false, bv, (short)0, o2, false, false);
      p = vbase + (size_t)(48 + l15) * T_ + j0 + hi * 16;
      #pragma unroll
      for (int i = 0; i < 16; ++i) bv[i] = p[i];
      o3 = __builtin_amdgcn_wmma_f32_16x16x32_bf16(false, ap, false, bv, (short)0, o3, false, false);
    }
  }

  // ---- normalize and store: out (B,T,C) bf16, col = h*64 + dt*16 + l15 ----
  __bf16* po = O + ((size_t)(b * T_ + q0 + hi * 8)) * C_ + h * D_ + l15;
  #pragma unroll
  for (int r = 0; r < 8; ++r) {
    const float il = 1.0f / lrow[r];
    po[(size_t)r * C_ + 0]  = (__bf16)(o0[r] * il);
    po[(size_t)r * C_ + 16] = (__bf16)(o1[r] * il);
    po[(size_t)r * C_ + 32] = (__bf16)(o2[r] * il);
    po[(size_t)r * C_ + 48] = (__bf16)(o3[r] * il);
  }
}

// ---------------------------------------------------------------------------

extern "C" void kernel_launch(void* const* d_in, const int* in_sizes, int n_in,
                              void* d_out, int out_size, void* d_ws, size_t ws_size,
                              hipStream_t stream) {
  const float* x    = (const float*)d_in[0];
  const float* Wq   = (const float*)d_in[1];
  const float* Wk   = (const float*)d_in[2];
  const float* Wv   = (const float*)d_in[3];
  const float* Wo   = (const float*)d_in[4];
  const float* qn_w = (const float*)d_in[5];
  const float* kn_w = (const float*)d_in[6];

  const int BTC = B_ * T_ * C_;   // 4194304
  const int CC  = C_ * C_;        // 1048576

  // ---- carve workspace ----
  char* w = (char*)d_ws;
  size_t off = 0;
  auto take = [&](size_t bytes) { void* p = w + off; off += (bytes + 255) & ~(size_t)255; return p; };
  __bf16* xb  = (__bf16*)take((size_t)BTC * 2);
  __bf16* wqb = (__bf16*)take((size_t)CC * 2);
  __bf16* wkb = (__bf16*)take((size_t)CC * 2);
  __bf16* wvb = (__bf16*)take((size_t)CC * 2);
  __bf16* wob = (__bf16*)take((size_t)CC * 2);
  float*  qf  = (float*) take((size_t)BTC * 4);
  float*  kf  = (float*) take((size_t)BTC * 4);
  float*  vf  = (float*) take((size_t)BTC * 4);
  __bf16* qb  = (__bf16*)take((size_t)BTC * 2);  // (B,H,T,D)
  __bf16* kb  = (__bf16*)take((size_t)BTC * 2);  // (B,H,T,D)
  __bf16* vtb = (__bf16*)take((size_t)BTC * 2);  // (B,H,D,T)
  __bf16* ab  = (__bf16*)take((size_t)BTC * 2);  // attention out (B,T,C)

  // ---- 1) bf16 conversions ----
  f32_to_bf16<<<2048, 256, 0, stream>>>(x, xb, BTC);
  f32_to_bf16<<<1024, 256, 0, stream>>>(Wq, wqb, CC);
  f32_to_bf16<<<1024, 256, 0, stream>>>(Wk, wkb, CC);
  f32_to_bf16<<<1024, 256, 0, stream>>>(Wv, wvb, CC);
  f32_to_bf16<<<1024, 256, 0, stream>>>(Wo, wob, CC);

  // ---- 2) Q/K/V projections: (4096x1024) @ (1024x1024)^T ----
  dim3 ggrid(B_ * T_ / 128, C_ / 128);
  gemm_bf16_wmma<<<ggrid, 256, 0, stream>>>(xb, wqb, qf, B_ * T_, C_, C_);
  gemm_bf16_wmma<<<ggrid, 256, 0, stream>>>(xb, wkb, kf, B_ * T_, C_, C_);
  gemm_bf16_wmma<<<ggrid, 256, 0, stream>>>(xb, wvb, vf, B_ * T_, C_, C_);

  // ---- 3) RMSNorm+RoPE (q,k) and V transpose ----
  const int rows = B_ * T_ * H_;  // 65536, 8 rows per block
  norm_rope_kernel<<<rows / 8, 256, 0, stream>>>(qf, qn_w, qb);
  norm_rope_kernel<<<rows / 8, 256, 0, stream>>>(kf, kn_w, kb);
  v_transpose_kernel<<<BTC / 256, 256, 0, stream>>>(vf, vtb);

  // ---- 4) flash attention ----
  dim3 agrid(T_ / 128, B_ * H_);
  flash_attn_kernel<<<agrid, 256, 0, stream>>>(qb, kb, vtb, ab);

  // ---- 5) output projection into f32 d_out ----
  gemm_bf16_wmma<<<ggrid, 256, 0, stream>>>(ab, wob, (float*)d_out, B_ * T_, C_, C_);
}
